// AtomAttentionEncoder_33346126086472
// MI455X (gfx1250) — matrix operand order repaired and checked by reading, not compiled, not run on hardware
//
#include <hip/hip_runtime.h>
#include <hip/hip_bf16.h>
#include <math.h>

// ---------------------------------------------------------------------------
// AtomAttentionEncoder for MI455X (gfx1250, wave32, WMMA 16x16x32 f16).
// Roofline: ~5 GFLOP vs ~150MB traffic @ 23.3TB/s -> memory bound; f16 WMMA
// w/ f32 accum keeps matrix math far off the critical path. token_pairs
// (67MB) lives in the 192MB L2 across the 512 window blocks.
// ---------------------------------------------------------------------------

typedef _Float16 half_t;
typedef __attribute__((ext_vector_type(16))) _Float16 v16h;
typedef __attribute__((ext_vector_type(8)))  float    v8f;

#define NATOMS 16384
#define CATOM  128
#define NTOK   1024
#define NHEADS 4
#define NQ     32
#define NK     128
#define DHEAD  32
#define NBLK   (NATOMS / NQ)       // 512
#define PAD    48                  // N_K/2 - N_Q/2
#define INV_SQRT_D 0.17677669529663689f
#define INFB   1.0e8f

// ----- WMMA fragment helpers (A/B k contiguous per lane half, C/D per ISA, wave32) -------------

static __device__ inline v8f wmma16(v16h a, v16h b, v8f c) {
  return __builtin_amdgcn_wmma_f32_16x16x32_f16(false, a, false, b, (short)0, c,
                                                false, false);
}

// A (16x32 f16): lane L row=L%16.. ; halfs[0..15]=K[hi*16+0..15]
static __device__ inline v16h frag_a(const half_t* A, int ld, int row0, int kb,
                                     int lane) {
  int r = lane & 15, hi = lane >> 4;
  const half_t* p = A + (size_t)(row0 + r) * ld + kb + hi * 16;
  v16h f;
#pragma unroll
  for (int j = 0; j < 16; ++j) f[j] = p[j];
  return f;
}

// B (32x16 f16) from source S[col][k] (k contiguous): B[k][n] = S[n][koff+k]
static __device__ inline v16h frag_b_kc(const half_t* S, int ld, int col0,
                                        int koff, int lane) {
  int c = lane & 15, hi = lane >> 4;
  const half_t* p = S + (size_t)(col0 + c) * ld + koff + hi * 16;
  v16h f;
#pragma unroll
  for (int j = 0; j < 16; ++j) f[j] = p[j];
  return f;
}

// B (32x16 f16) from row-major S[k][n]
static __device__ inline v16h frag_b_rm(const half_t* S, int ld, int kb,
                                        int col0, int lane) {
  int c = lane & 15, hi = lane >> 4;
  const half_t* p = S + (size_t)(kb + hi * 16) * ld + col0 + c;
  v16h f;
#pragma unroll
  for (int j = 0; j < 16; ++j) f[j] = p[(size_t)j * ld];
  return f;
}

// B (32x16 f16) from row-major f32 source (converted on load)
static __device__ inline v16h frag_b_rm_f32(const float* S, int ld, int kb,
                                            int col0, int lane) {
  int c = lane & 15, hi = lane >> 4;
  const float* p = S + (size_t)(kb + hi * 16) * ld + col0 + c;
  v16h f;
#pragma unroll
  for (int j = 0; j < 16; ++j) f[j] = (half_t)p[(size_t)j * ld];
  return f;
}

// C/D (16x16 f32): f[i] = C[m0 + hi*8 + i][n0 + lane%16]
static __device__ inline v8f frag_c_load(const float* C, int ld, int m0, int n0,
                                         int lane) {
  int c = lane & 15, hi = lane >> 4;
  v8f f;
#pragma unroll
  for (int i = 0; i < 8; ++i) f[i] = C[(size_t)(m0 + hi * 8 + i) * ld + n0 + c];
  return f;
}

static __device__ inline void frag_c_store(float* C, int ld, int m0, int n0,
                                           int lane, v8f f) {
  int c = lane & 15, hi = lane >> 4;
#pragma unroll
  for (int i = 0; i < 8; ++i) C[(size_t)(m0 + hi * 8 + i) * ld + n0 + c] = f[i];
}

// ----- Kernel 0: convert Wo / W_op to f16 ------------------------------------

__global__ void __launch_bounds__(256) prep_kernel(const float* __restrict__ Wo,
                                                   const float* __restrict__ Wop,
                                                   half_t* __restrict__ woh,
                                                   half_t* __restrict__ woph) {
  int i = blockIdx.x * 256 + threadIdx.x;
  if (i < CATOM * CATOM) {
    woh[i]  = (half_t)Wo[i];
    woph[i] = (half_t)Wop[i];
  }
}

// ----- Kernel 1: zero token sums + counts ------------------------------------

__global__ void __launch_bounds__(256) zero_kernel(float* __restrict__ sums,
                                                   float* __restrict__ cnt) {
  int i = blockIdx.x * 256 + threadIdx.x;
  if (i < NTOK * CATOM) sums[i] = 0.0f;
  if (i < NTOK) cnt[i] = 0.0f;
}

// ----- Kernel 2: segment counts ---------------------------------------------

__global__ void __launch_bounds__(256) count_kernel(const int* __restrict__ tok_idx,
                                                    float* __restrict__ cnt) {
  int i = blockIdx.x * 256 + threadIdx.x;
  if (i < NATOMS) atomicAdd(&cnt[tok_idx[i]], 1.0f);
}

// ----- Kernel 3: q/k/v/g projections (WMMA) ----------------------------------
// X[16384x128] x W[128x128] for each of Wq,Wk,Wv,Wg. One block = 16 atoms,
// 8 waves x 4 N-tiles = 32 N-tiles covering the 512 combined output columns.

__global__ void __launch_bounds__(256) proj_kernel(
    const float* __restrict__ X, const float* __restrict__ Wq,
    const float* __restrict__ Wk, const float* __restrict__ Wv,
    const float* __restrict__ Wg, half_t* __restrict__ qh,
    half_t* __restrict__ kh, half_t* __restrict__ vh,
    float* __restrict__ gbuf) {
  __shared__ half_t Alds[16 * CATOM];
  int tid = threadIdx.x;
  int m0 = blockIdx.x * 16;
  for (int i = tid; i < 16 * CATOM; i += 256)
    Alds[i] = (half_t)X[(size_t)(m0 + (i >> 7)) * CATOM + (i & 127)];
  __syncthreads();

  int wave = tid >> 5, lane = tid & 31;
  const float* Ws[4] = {Wq, Wk, Wv, Wg};
  half_t* Outs[3] = {qh, kh, vh};

#pragma unroll
  for (int t = 0; t < 4; ++t) {
    int nt = wave * 4 + t;            // 0..31
    int which = nt >> 3;              // 0..3 -> q,k,v,g
    int col0 = (nt & 7) * 16;         // 0..112
    const float* W = Ws[which];
    v8f acc = {0.f, 0.f, 0.f, 0.f, 0.f, 0.f, 0.f, 0.f};
#pragma unroll
    for (int kc = 0; kc < 4; ++kc) {
      v16h a = frag_a(Alds, CATOM, 0, kc * 32, lane);
      v16h b = frag_b_rm_f32(W, CATOM, kc * 32, col0, lane);
      acc = wmma16(a, b, acc);
    }
    int c = lane & 15, hi = lane >> 4;
    if (which < 3) {
      half_t* O = Outs[which];
#pragma unroll
      for (int i = 0; i < 8; ++i) {
        int m = hi * 8 + i;
        O[(size_t)(m0 + m) * CATOM + col0 + c] = (half_t)acc[i];
      }
    } else {
#pragma unroll
      for (int i = 0; i < 8; ++i) {
        int m = hi * 8 + i;
        gbuf[(size_t)(m0 + m) * CATOM + col0 + c] =
            1.0f / (1.0f + __expf(-acc[i]));
      }
    }
  }
}

// ----- Kernel 4: windowed attention per 32-atom block ------------------------
// 512 blocks x 256 threads (8 waves). Dynamic LDS ~200KB (<=320KB/WGP).

__global__ void __launch_bounds__(256) attn_kernel(
    const half_t* __restrict__ qh, const half_t* __restrict__ kh,
    const half_t* __restrict__ vh, const float* __restrict__ gbuf,
    const float* __restrict__ token_pairs, const int* __restrict__ tok_idx,
    const float* __restrict__ mask, const float* __restrict__ ln_scale,
    const float* __restrict__ ln_bias, const float* __restrict__ W_pair,
    const half_t* __restrict__ woh, const half_t* __restrict__ woph,
    const float* __restrict__ b_op, float* __restrict__ sums) {
  extern __shared__ char smem_raw[];
  half_t* Qs  = (half_t*)smem_raw;        // 32x128
  half_t* Ks  = Qs + NQ * CATOM;          // 128x128
  half_t* Vs  = Ks + NK * CATOM;          // 128x128
  half_t* Ps  = Vs + NK * CATOM;          // 4x32x128 probs
  half_t* Os  = Ps + NHEADS * NQ * NK;    // 32x128 (g*o)
  half_t* AOh = Os + NQ * CATOM;          // 32x128 f16 attn_out
  float*  Sb  = (float*)(AOh + NQ * CATOM); // 4x32x128 f32 scores
  float*  AOf = Sb + NHEADS * NQ * NK;    // 32x128 f32 attn_out

  __shared__ int   tokl[NQ];
  __shared__ int   tokm[NK];
  __shared__ int   vk[NK];
  __shared__ float mbias[NK];

  int n = blockIdx.x;
  int tid = threadIdx.x;
  int a0 = n * NQ;
  int wave = tid >> 5, lane = tid & 31;

  // ---- Phase A: stage Q (pre-scaled), K/V windows, token indices ----
  for (int i = tid; i < NQ * CATOM; i += 256)
    Qs[i] = (half_t)((float)qh[(size_t)(a0 + (i >> 7)) * CATOM + (i & 127)] *
                     INV_SQRT_D);
  for (int i = tid; i < NK * CATOM; i += 256) {
    int k = i >> 7, c = i & 127;
    int pos = a0 + k - PAD;
    bool valid = (pos >= 0) && (pos < NATOMS);
    int pc = valid ? pos : 0;
    Ks[i] = valid ? kh[(size_t)pc * CATOM + c] : (half_t)0.0f;
    Vs[i] = valid ? vh[(size_t)pc * CATOM + c] : (half_t)0.0f;
  }
  if (tid < NQ) tokl[tid] = tok_idx[a0 + tid];
  if (tid < NK) {
    int pos = a0 + tid - PAD;
    bool valid = (pos >= 0) && (pos < NATOMS);
    vk[tid] = valid ? 1 : 0;
    tokm[tid] = valid ? tok_idx[pos] : 0;
    mbias[tid] = valid ? INFB * (mask[pos] - 1.0f) : -INFB;
  }
  __syncthreads();

  // ---- Phase B: pair bias (LayerNorm over 16ch, dot W_pair) + mask bias ----
  for (int idx = tid; idx < NQ * NK; idx += 256) {
    int q = idx >> 7, k = idx & 127;
    float base = mbias[k];
    if (!vk[k]) {
#pragma unroll
      for (int h = 0; h < NHEADS; ++h) Sb[h * (NQ * NK) + idx] = base;
    } else {
      const float* p =
          token_pairs + ((size_t)tokl[q] * NTOK + tokm[k]) * 16;
      float x[16], s = 0.0f;
#pragma unroll
      for (int c = 0; c < 16; ++c) { x[c] = p[c]; s += x[c]; }
      float mu = s * (1.0f / 16.0f);
      float ss = 0.0f;
#pragma unroll
      for (int c = 0; c < 16; ++c) { float d = x[c] - mu; ss += d * d; }
      float rstd = rsqrtf(ss * (1.0f / 16.0f) + 1e-5f);
      float b[4] = {0.f, 0.f, 0.f, 0.f};
#pragma unroll
      for (int c = 0; c < 16; ++c) {
        float xn = (x[c] - mu) * rstd * ln_scale[c] + ln_bias[c];
#pragma unroll
        for (int h = 0; h < NHEADS; ++h) b[h] += xn * W_pair[c * 4 + h];
      }
#pragma unroll
      for (int h = 0; h < NHEADS; ++h) Sb[h * (NQ * NK) + idx] = b[h] + base;
    }
  }
  __syncthreads();

  // ---- Phase C: scores = (Q/sqrt d) K^T + bias.  64 WMMA tile-jobs ----
#pragma unroll
  for (int ji = 0; ji < 8; ++ji) {
    int j = wave * 8 + ji;                 // 0..63
    int h = j >> 4, rem = j & 15;
    int mt = rem >> 3, nt = rem & 7;       // mt: q tile, nt: key tile
    v16h a = frag_a(Qs, CATOM, mt * 16, h * DHEAD, lane);
    v16h b = frag_b_kc(Ks, CATOM, nt * 16, h * DHEAD, lane);
    v8f cfr = frag_c_load(Sb + h * (NQ * NK), NK, mt * 16, nt * 16, lane);
    v8f d = wmma16(a, b, cfr);
    frag_c_store(Sb + h * (NQ * NK), NK, mt * 16, nt * 16, lane, d);
  }
  __syncthreads();

  // ---- Phase D: softmax over 128 keys; probs -> f16 ----
  if (tid < NHEADS * NQ) {
    float* row = Sb + (tid >> 5) * (NQ * NK) + (tid & 31) * NK;
    half_t* pr = Ps + (tid >> 5) * (NQ * NK) + (tid & 31) * NK;
    float m = -3.0e38f;
    for (int k = 0; k < NK; ++k) m = fmaxf(m, row[k]);
    float s = 0.0f;
    for (int k = 0; k < NK; ++k) { float e = __expf(row[k] - m); row[k] = e; s += e; }
    float inv = 1.0f / s;
    for (int k = 0; k < NK; ++k) pr[k] = (half_t)(row[k] * inv);
  }
  __syncthreads();

  // ---- Phase E: O = P x V, gated by g.  16 jobs x 4 chained WMMA ----
#pragma unroll
  for (int ji = 0; ji < 2; ++ji) {
    int j = wave * 2 + ji;                 // 0..15
    int h = j >> 2, rem = j & 3;
    int mt = rem >> 1, dt = rem & 1;
    v8f acc = {0.f, 0.f, 0.f, 0.f, 0.f, 0.f, 0.f, 0.f};
#pragma unroll
    for (int kc = 0; kc < 4; ++kc) {
      v16h a = frag_a(Ps + h * (NQ * NK), NK, mt * 16, kc * 32, lane);
      v16h b = frag_b_rm(Vs, CATOM, kc * 32, h * DHEAD + dt * 16, lane);
      acc = wmma16(a, b, acc);
    }
    int c = lane & 15, hi = lane >> 4;
#pragma unroll
    for (int i = 0; i < 8; ++i) {
      int m = mt * 16 + hi * 8 + i;
      int col = h * DHEAD + dt * 16 + c;
      float gv = gbuf[(size_t)(a0 + m) * CATOM + col];
      Os[m * CATOM + col] = (half_t)(acc[i] * gv);
    }
  }
  __syncthreads();

  // ---- Phase F: attn_out = (g*o) x Wo  (32x128x128) ----
#pragma unroll
  for (int ji = 0; ji < 2; ++ji) {
    int j = wave * 2 + ji;
    int mt = j >> 3, nt = j & 7;
    v8f acc = {0.f, 0.f, 0.f, 0.f, 0.f, 0.f, 0.f, 0.f};
#pragma unroll
    for (int kc = 0; kc < 4; ++kc) {
      v16h a = frag_a(Os, CATOM, mt * 16, kc * 32, lane);
      v16h b = frag_b_rm(woh, CATOM, kc * 32, nt * 16, lane);
      acc = wmma16(a, b, acc);
    }
    int c = lane & 15, hi = lane >> 4;
#pragma unroll
    for (int i = 0; i < 8; ++i) {
      int m = mt * 16 + hi * 8 + i;
      int col = nt * 16 + c;
      AOf[m * CATOM + col] = acc[i];
      AOh[m * CATOM + col] = (half_t)acc[i];
    }
  }
  __syncthreads();

  // ---- Phase G: atom_out = sigmoid(attn_out @ W_op + b) * attn_out,
  //               scattered into token sums (global_atomic_add_f32) ----
#pragma unroll
  for (int ji = 0; ji < 2; ++ji) {
    int j = wave * 2 + ji;
    int mt = j >> 3, nt = j & 7;
    v8f acc = {0.f, 0.f, 0.f, 0.f, 0.f, 0.f, 0.f, 0.f};
#pragma unroll
    for (int kc = 0; kc < 4; ++kc) {
      v16h a = frag_a(AOh, CATOM, mt * 16, kc * 32, lane);
      v16h b = frag_b_rm(woph, CATOM, kc * 32, nt * 16, lane);
      acc = wmma16(a, b, acc);
    }
    int c = lane & 15, hi = lane >> 4;
#pragma unroll
    for (int i = 0; i < 8; ++i) {
      int m = mt * 16 + hi * 8 + i;
      int col = nt * 16 + c;
      float u = acc[i] + b_op[col];
      float sg = 1.0f / (1.0f + __expf(-u));
      float val = sg * AOf[m * CATOM + col];
      atomicAdd(&sums[(size_t)tokl[m] * CATOM + col], val);
    }
  }
}

// ----- Kernel 5: finalize token mean ----------------------------------------

__global__ void __launch_bounds__(256) final_kernel(const float* __restrict__ sums,
                                                    const float* __restrict__ cnt,
                                                    float* __restrict__ out) {
  int i = blockIdx.x * 256 + threadIdx.x;
  if (i < NTOK * CATOM) out[i] = sums[i] / fmaxf(cnt[i >> 7], 1.0f);
}

// ----- host-side launch ------------------------------------------------------

extern "C" void kernel_launch(void* const* d_in, const int* in_sizes, int n_in,
                              void* d_out, int out_size, void* d_ws,
                              size_t ws_size, hipStream_t stream) {
  const float* atom_single = (const float*)d_in[0];
  const float* token_pairs = (const float*)d_in[1];
  const int*   tok_idx     = (const int*)d_in[2];
  const float* mask        = (const float*)d_in[3];
  // d_in[4] = n_tokens (fixed at 1024)
  const float* Wq       = (const float*)d_in[5];
  const float* Wk       = (const float*)d_in[6];
  const float* Wv       = (const float*)d_in[7];
  const float* Wg       = (const float*)d_in[8];
  const float* Wo       = (const float*)d_in[9];
  const float* ln_scale = (const float*)d_in[10];
  const float* ln_bias  = (const float*)d_in[11];
  const float* W_pair   = (const float*)d_in[12];
  const float* W_op     = (const float*)d_in[13];
  const float* b_op     = (const float*)d_in[14];
  float* out = (float*)d_out;

  char* ws = (char*)d_ws;
  size_t off = 0;
  auto carve = [&](size_t bytes) -> void* {
    void* p = ws + off;
    off = (off + bytes + 255) & ~(size_t)255;
    return p;
  };
  half_t* qh   = (half_t*)carve((size_t)NATOMS * CATOM * sizeof(half_t));
  half_t* kh   = (half_t*)carve((size_t)NATOMS * CATOM * sizeof(half_t));
  half_t* vh   = (half_t*)carve((size_t)NATOMS * CATOM * sizeof(half_t));
  half_t* woh  = (half_t*)carve((size_t)CATOM * CATOM * sizeof(half_t));
  half_t* woph = (half_t*)carve((size_t)CATOM * CATOM * sizeof(half_t));
  float*  gbuf = (float*)carve((size_t)NATOMS * CATOM * sizeof(float));
  float*  sums = (float*)carve((size_t)NTOK * CATOM * sizeof(float));
  float*  cnt  = (float*)carve((size_t)NTOK * sizeof(float));

  prep_kernel<<<dim3((CATOM * CATOM + 255) / 256), dim3(256), 0, stream>>>(
      Wo, W_op, woh, woph);
  zero_kernel<<<dim3((NTOK * CATOM + 255) / 256), dim3(256), 0, stream>>>(sums,
                                                                          cnt);
  count_kernel<<<dim3((NATOMS + 255) / 256), dim3(256), 0, stream>>>(tok_idx,
                                                                     cnt);
  proj_kernel<<<dim3(NATOMS / 16), dim3(256), 0, stream>>>(
      atom_single, Wq, Wk, Wv, Wg, qh, kh, vh, gbuf);

  // dynamic LDS: Q 8K + K 32K + V 32K + P 32K + O 8K + AOh 8K + Sb 64K + AOf 16K
  size_t smem = (size_t)(NQ * CATOM + 2 * NK * CATOM + NHEADS * NQ * NK +
                         2 * NQ * CATOM) * sizeof(half_t) +
                (size_t)(NHEADS * NQ * NK + NQ * CATOM) * sizeof(float);
  attn_kernel<<<dim3(NBLK), dim3(256), smem, stream>>>(
      qh, kh, vh, gbuf, token_pairs, tok_idx, mask, ln_scale, ln_bias, W_pair,
      woh, woph, b_op, sums);

  final_kernel<<<dim3((NTOK * CATOM + 255) / 256), dim3(256), 0, stream>>>(
      sums, cnt, out);
}
